// LeikaGPT_59399397704256
// MI455X (gfx1250) — matrix operand order repaired and checked by reading, not compiled, not run on hardware
//
#include <hip/hip_runtime.h>
#include <hip/hip_bf16.h>

// ---------------------------------------------------------------------------
// GPT-2 small forward for MI455X (gfx1250), wave32 + WMMA f32_16x16x32_f16.
// Model: V=50257, D=768, L=12, H=12, HD=64, B=2, T=1024.
// ---------------------------------------------------------------------------

#define GPT_B 2
#define GPT_T 1024
#define GPT_D 768
#define GPT_L 12
#define GPT_H 12
#define GPT_HD 64
#define GPT_V 50257
#define GPT_FF (4 * GPT_D)
#define NTOK (GPT_B * GPT_T)     // 2048 rows in every GEMM (multiple of 128)
#define LN_EPS 1e-5f

typedef __attribute__((ext_vector_type(16))) _Float16 v16h;
typedef __attribute__((ext_vector_type(2)))  __fp16   v2fp16;
typedef __attribute__((ext_vector_type(8)))  float    v8f;

union FragU {
    unsigned u[8];
    uint4    q[2];
    v16h     h;
};

// Single-instruction f32 pair -> packed f16 dword (v_cvt_pk_rtz_f16_f32).
__device__ __forceinline__ unsigned pk2h(float a, float b) {
    union { v2fp16 h; unsigned u; } r;
    r.h = __builtin_amdgcn_cvt_pkrtz(a, b);
    return r.u;
}

__device__ __forceinline__ v8f wmma32(const FragU& a, const FragU& b, v8f c) {
    // D = A(16x32 f16) * B(32x16 f16) + C(16x16 f32)
    return __builtin_amdgcn_wmma_f32_16x16x32_f16(
        false, a.h, false, b.h, (short)0, c, false, false);
}

__device__ __forceinline__ v8f vzero8() {
    v8f z = {0.f, 0.f, 0.f, 0.f, 0.f, 0.f, 0.f, 0.f};
    return z;
}

// A-fragment (16x32, row-major source, fp32 -> f16 pack).
// Layout: lane row = lane&15; hh = lane>>4;
//   VGPR v : K = 16*hh + 2v, 2v+1 (contiguous 16 halves, same as B)
// Loads batched first so they pipeline; converts afterwards.
__device__ __forceinline__ FragU frag_a_rm(const float* base, int ld, int lr, int hh) {
    const float* p = base + (size_t)lr * ld + hh * 16;
    float2 t[8];
#pragma unroll
    for (int v = 0; v < 4; ++v) t[v]     = *(const float2*)(p + 2 * v);
#pragma unroll
    for (int v = 0; v < 4; ++v) t[4 + v] = *(const float2*)(p + 8 + 2 * v);
    FragU f;
#pragma unroll
    for (int v = 0; v < 8; ++v) f.u[v] = pk2h(t[v].x, t[v].y);
    return f;
}

// B-fragment (32x16). Source is [n][k] row-major (K contiguous per column n).
// Layout used here: lane col = lane&15; K = 16*hh + 2v, 2v+1 (contiguous 16 halves).
__device__ __forceinline__ FragU frag_b_rm(const float* base, int ld, int lr, int hh) {
    const float* p = base + (size_t)lr * ld + hh * 16;
    float2 t[8];
#pragma unroll
    for (int v = 0; v < 8; ++v) t[v] = *(const float2*)(p + 2 * v);
    FragU f;
#pragma unroll
    for (int v = 0; v < 8; ++v) f.u[v] = pk2h(t[v].x, t[v].y);
    return f;
}

// B-fragment from a [k][n] source, e.g. V[T][HD].
__device__ __forceinline__ FragU frag_b_cm(const float* base, int ld, int lr, int hh) {
    const float* p = base + (size_t)(hh * 16) * ld + lr;
    float t0[8], t1[8];
#pragma unroll
    for (int v = 0; v < 8; ++v) {
        t0[v] = p[(2 * v) * ld];
        t1[v] = p[(2 * v + 1) * ld];
    }
    FragU f;
#pragma unroll
    for (int v = 0; v < 8; ++v) f.u[v] = pk2h(t0[v], t1[v]);
    return f;
}

// ---------------------------------------------------------------------------
// Embedding: x[b,t,:] = tok_emb[idx[b,t],:] + pos_emb[t,:]
// ---------------------------------------------------------------------------
__global__ __launch_bounds__(256) void embed_kernel(
    const int* __restrict__ idx, const float* __restrict__ tok,
    const float* __restrict__ pos, float* __restrict__ x) {
    const int row = blockIdx.x;                 // 0..NTOK-1
    const int t   = row & (GPT_T - 1);
    const int id  = idx[row];
    const float* te = tok + (size_t)id * GPT_D;
    const float* pe = pos + (size_t)t * GPT_D;
    float* xr = x + (size_t)row * GPT_D;
    for (int c = threadIdx.x; c < GPT_D; c += 256) xr[c] = te[c] + pe[c];
}

// ---------------------------------------------------------------------------
// LayerNorm: one row (768) per 256-thread block.
// ---------------------------------------------------------------------------
__global__ __launch_bounds__(256) void ln_kernel(
    const float* __restrict__ X, const float* __restrict__ g,
    const float* __restrict__ b, float* __restrict__ O) {
    const int row = blockIdx.x;
    const float* xr = X + (size_t)row * GPT_D;
    float v0[3], s = 0.f, ss = 0.f;
#pragma unroll
    for (int i = 0; i < 3; ++i) {
        float v = xr[threadIdx.x + 256 * i];
        v0[i] = v; s += v; ss += v * v;
    }
#pragma unroll
    for (int o = 16; o; o >>= 1) {
        s  += __shfl_xor(s,  o, 32);
        ss += __shfl_xor(ss, o, 32);
    }
    __shared__ float rs[8], rss[8];
    const int w = threadIdx.x >> 5, lane = threadIdx.x & 31;
    if (lane == 0) { rs[w] = s; rss[w] = ss; }
    __syncthreads();
    float ts = 0.f, tss = 0.f;
#pragma unroll
    for (int i = 0; i < 8; ++i) { ts += rs[i]; tss += rss[i]; }
    const float mean = ts * (1.f / GPT_D);
    const float var  = tss * (1.f / GPT_D) - mean * mean;
    const float inv  = rsqrtf(var + LN_EPS);
    float* orow = O + (size_t)row * GPT_D;
#pragma unroll
    for (int i = 0; i < 3; ++i) {
        int c = threadIdx.x + 256 * i;
        orow[c] = (v0[i] - mean) * inv * g[c] + b[c];
    }
}

// ---------------------------------------------------------------------------
// WMMA GEMM: C[M,N] = A[M,K] @ B[K,N] (+bias)(+GELU)(+residual)
// Compile-time epilogue flags -> straight-line epilogue code.
// Block: 256 threads (8 waves). Block tile 128x128, wave tile 32x64 (2x4
// WMMA tiles), BK=32. M must be a multiple of 128 (here always 2048).
// ---------------------------------------------------------------------------
template <bool HAS_BIAS, bool HAS_RES, bool GELU>
__global__ __launch_bounds__(256) void gemm_wmma(
    const float* __restrict__ A, const float* __restrict__ Bm,
    const float* __restrict__ bias, const float* __restrict__ R,
    float* __restrict__ C, int N, int K) {
    __shared__ unsigned As[128][16];   // 128 rows x 32 f16 (pairs as dwords)
    __shared__ _Float16 Bs[128][32];   // [n][k] transposed tile

    const int tid  = threadIdx.x;
    const int lane = tid & 31, w = tid >> 5;
    const int hh   = lane >> 4, lr = lane & 15;
    const int wm   = (w & 3) * 32;       // 4 waves along M
    const int wn   = (w >> 2) * 64;      // 2 waves along N
    const int m0   = blockIdx.y * 128, n0 = blockIdx.x * 128;

    v8f acc[2][4];
#pragma unroll
    for (int mt = 0; mt < 2; ++mt)
#pragma unroll
        for (int nt = 0; nt < 4; ++nt) acc[mt][nt] = vzero8();

    for (int k0 = 0; k0 < K; k0 += 32) {
        // --- stage A tile (128x32 fp32): batch loads, then convert/store ---
        float4 fa[4];
#pragma unroll
        for (int i = 0; i < 4; ++i) {
            int c   = tid + i * 256;
            int row = c >> 3;
            int kq  = (c & 7) * 4;
            fa[i] = *(const float4*)(A + (size_t)(m0 + row) * K + k0 + kq);
        }
        // --- stage B tile (32x128, clamped column): batch loads ---
        float fb[16];
#pragma unroll
        for (int i = 0; i < 16; ++i) {
            int c  = tid + i * 256;
            int kk = c >> 7;
            int n  = c & 127;
            int gn = n0 + n;
            if (gn >= N) gn = N - 1;   // clamp: padding cols never written back
            fb[i] = Bm[(size_t)(k0 + kk) * N + gn];
        }
#pragma unroll
        for (int i = 0; i < 4; ++i) {
            int c   = tid + i * 256;
            int row = c >> 3;
            int kq  = (c & 7) * 4;
            As[row][(kq >> 1)]     = pk2h(fa[i].x, fa[i].y);
            As[row][(kq >> 1) + 1] = pk2h(fa[i].z, fa[i].w);
        }
#pragma unroll
        for (int i = 0; i < 16; ++i) {
            int c  = tid + i * 256;
            int kk = c >> 7;
            int n  = c & 127;
            Bs[n][kk] = (_Float16)fb[i];
        }
        __syncthreads();

        FragU a[2], bf[4];
#pragma unroll
        for (int mt = 0; mt < 2; ++mt) {
            int r = wm + mt * 16 + lr;
            a[mt].q[0] = *(const uint4*)&As[r][hh * 8];
            a[mt].q[1] = *(const uint4*)&As[r][hh * 8 + 4];
        }
#pragma unroll
        for (int nt = 0; nt < 4; ++nt) {
            int n = wn + nt * 16 + lr;
            const uint4* bp = (const uint4*)&Bs[n][hh * 16];
            bf[nt].q[0] = bp[0];
            bf[nt].q[1] = bp[1];
        }
#pragma unroll
        for (int mt = 0; mt < 2; ++mt)
#pragma unroll
            for (int nt = 0; nt < 4; ++nt)
                acc[mt][nt] = wmma32(a[mt], bf[nt], acc[mt][nt]);
        __syncthreads();
    }

    // --- epilogue: straight-line (flags are compile-time) ---
#pragma unroll
    for (int mt = 0; mt < 2; ++mt)
#pragma unroll
        for (int nt = 0; nt < 4; ++nt) {
            int gc = n0 + wn + nt * 16 + lr;
            if (gc >= N) continue;
            float bv = HAS_BIAS ? bias[gc] : 0.f;
            const int gr0 = m0 + wm + mt * 16 + 8 * hh;
            float res[8];
            if (HAS_RES) {
#pragma unroll
                for (int v = 0; v < 8; ++v)
                    res[v] = R[(size_t)(gr0 + v) * N + gc];
            }
#pragma unroll
            for (int v = 0; v < 8; ++v) {
                float o = acc[mt][nt][v] + bv;
                if (GELU) o = 0.5f * o * (1.f + erff(o * 0.70710678118654752f));
                if (HAS_RES) o += res[v];
                C[(size_t)(gr0 + v) * N + gc] = o;
            }
        }
}

// ---------------------------------------------------------------------------
// Attention: one block per (b, head, 64-query tile). Scores kept in 256 KB of
// dynamic LDS (CDNA5 WGP has 320 KB). Q,K,V,Y are [B,T,D] with head offset.
// ---------------------------------------------------------------------------
__global__ __launch_bounds__(256) void attn_wmma(
    const float* __restrict__ Q, const float* __restrict__ Kb,
    const float* __restrict__ V, float* __restrict__ Y) {
    extern __shared__ float S[];                  // [64][1024] fp32 scores
    const int qt = blockIdx.x;                    // 0..15 (64-row query tiles)
    const int hI = blockIdx.y;                    // head
    const int b  = blockIdx.z;                    // batch
    const int tid = threadIdx.x, lane = tid & 31, w = tid >> 5;
    const int hh = lane >> 4, lr = lane & 15;

    const size_t headoff = (size_t)hI * GPT_HD;
    const float* qbase = Q + ((size_t)b * GPT_T + (size_t)qt * 64) * GPT_D + headoff;
    const float* kbase = Kb + (size_t)b * GPT_T * GPT_D + headoff;
    const float* vbase = V  + (size_t)b * GPT_T * GPT_D + headoff;
    float*       ybase = Y  + ((size_t)b * GPT_T + (size_t)qt * 64) * GPT_D + headoff;

    const int kmax = (qt + 1) * 64;               // causal: keys < kmax

    // ---- phase 1: S = Q @ K^T (64 x kmax), WMMA over HD=64 (2 chunks) ----
    const int ntn = kmax >> 4;                    // 16-col key tiles
    for (int t = w; t < 4 * ntn; t += 8) {
        int mt = t & 3, nt = t >> 2;
        v8f acc = vzero8();
#pragma unroll
        for (int c = 0; c < 2; ++c) {
            FragU a  = frag_a_rm(qbase + (size_t)(mt * 16) * GPT_D + c * 32, GPT_D, lr, hh);
            FragU bf = frag_b_rm(kbase + (size_t)(nt * 16) * GPT_D + c * 32, GPT_D, lr, hh);
            acc = wmma32(a, bf, acc);
        }
#pragma unroll
        for (int v = 0; v < 8; ++v)
            S[(mt * 16 + v + 8 * hh) * GPT_T + nt * 16 + lr] = acc[v];
    }
    __syncthreads();

    // ---- phase 2: causal softmax over rows of S (8 rows per wave) ----
    const float scale = 0.125f;                   // 1/sqrt(64)
    for (int rr = 0; rr < 8; ++rr) {
        int m  = w * 8 + rr;
        int qr = qt * 64 + m;
        float* row = S + (size_t)m * GPT_T;
        float mx = -3.4e38f;
        for (int i = lane; i < kmax; i += 32) {
            float s = (i <= qr) ? row[i] * scale : -3.4e38f;
            mx = fmaxf(mx, s);
        }
#pragma unroll
        for (int o = 16; o; o >>= 1) mx = fmaxf(mx, __shfl_xor(mx, o, 32));
        float sum = 0.f;
        for (int i = lane; i < kmax; i += 32) {
            float e = (i <= qr) ? __expf(row[i] * scale - mx) : 0.f;
            row[i] = e;
            sum += e;
        }
#pragma unroll
        for (int o = 16; o; o >>= 1) sum += __shfl_xor(sum, o, 32);
        float inv = 1.f / sum;
        for (int i = lane; i < kmax; i += 32) row[i] *= inv;
    }
    __syncthreads();

    // ---- phase 3: Y = P @ V (64x64 out = 16 tiles, 2 per wave) ----
    const int nchunks = kmax >> 5;
    for (int t2 = w * 2; t2 < w * 2 + 2; ++t2) {
        int mt = t2 & 3, nt = t2 >> 2;
        v8f acc = vzero8();
        for (int c = 0; c < nchunks; ++c) {
            FragU a  = frag_a_rm(S + (size_t)(mt * 16) * GPT_T + c * 32, GPT_T, lr, hh);
            FragU bf = frag_b_cm(vbase + (size_t)(c * 32) * GPT_D + nt * 16, GPT_D, lr, hh);
            acc = wmma32(a, bf, acc);
        }
#pragma unroll
        for (int v = 0; v < 8; ++v)
            ybase[(size_t)(mt * 16 + v + 8 * hh) * GPT_D + nt * 16 + lr] = acc[v];
    }
}

// ---------------------------------------------------------------------------
// Host launcher
// ---------------------------------------------------------------------------
extern "C" void kernel_launch(void* const* d_in, const int* in_sizes, int n_in,
                              void* d_out, int out_size, void* d_ws, size_t ws_size,
                              hipStream_t stream) {
    const int*   idx    = (const int*)d_in[0];
    const float* tok    = (const float*)d_in[1];
    const float* pos    = (const float*)d_in[2];
    const float* Wq     = (const float*)d_in[3];
    const float* Wk     = (const float*)d_in[4];
    const float* Wv     = (const float*)d_in[5];
    const float* Wo     = (const float*)d_in[6];
    const float* bo     = (const float*)d_in[7];
    const float* ln1_g  = (const float*)d_in[8];
    const float* ln1_b  = (const float*)d_in[9];
    const float* Wf1    = (const float*)d_in[10];
    const float* bf1    = (const float*)d_in[11];
    const float* Wf2    = (const float*)d_in[12];
    const float* bf2    = (const float*)d_in[13];
    const float* ln2_g  = (const float*)d_in[14];
    const float* ln2_b  = (const float*)d_in[15];
    const float* lnf_g  = (const float*)d_in[16];
    const float* lnf_b  = (const float*)d_in[17];
    const float* W_lm   = (const float*)d_in[18];
    float* out = (float*)d_out;

    // workspace layout (fp32)
    const size_t ACT = (size_t)NTOK * GPT_D;   // 1.57M floats
    float* ws = (float*)d_ws;
    float* x  = ws;
    float* h  = x + ACT;
    float* q  = h + ACT;
    float* k  = q + ACT;
    float* v  = k + ACT;
    float* y  = v + ACT;
    float* ff = y + ACT;                       // NTOK * 3072

    const dim3 blk(256);
    const dim3 gD((GPT_D + 127) / 128, NTOK / 128);    // N=768 GEMMs
    const dim3 gF((GPT_FF + 127) / 128, NTOK / 128);   // N=3072 GEMM
    const dim3 gV((GPT_V + 127) / 128, NTOK / 128);    // LM head
    const dim3 gAttn(GPT_T / 64, GPT_H, GPT_B);
    const size_t attn_lds = (size_t)64 * GPT_T * sizeof(float);  // 256 KB

    embed_kernel<<<NTOK, blk, 0, stream>>>(idx, tok, pos, x);

    for (int l = 0; l < GPT_L; ++l) {
        const size_t oDD = (size_t)l * GPT_D * GPT_D;
        const size_t oD  = (size_t)l * GPT_D;
        const size_t oDF = (size_t)l * GPT_D * GPT_FF;
        const size_t oF  = (size_t)l * GPT_FF;

        ln_kernel<<<NTOK, blk, 0, stream>>>(x, ln1_g + oD, ln1_b + oD, h);

        gemm_wmma<false, false, false><<<gD, blk, 0, stream>>>(
            h, Wq + oDD, nullptr, nullptr, q, GPT_D, GPT_D);
        gemm_wmma<false, false, false><<<gD, blk, 0, stream>>>(
            h, Wk + oDD, nullptr, nullptr, k, GPT_D, GPT_D);
        gemm_wmma<false, false, false><<<gD, blk, 0, stream>>>(
            h, Wv + oDD, nullptr, nullptr, v, GPT_D, GPT_D);

        attn_wmma<<<gAttn, blk, attn_lds, stream>>>(q, k, v, y);

        // x = x + y @ Wo + bo
        gemm_wmma<true, true, false><<<gD, blk, 0, stream>>>(
            y, Wo + oDD, bo + oD, x, x, GPT_D, GPT_D);

        ln_kernel<<<NTOK, blk, 0, stream>>>(x, ln2_g + oD, ln2_b + oD, h);

        // ff = gelu(h @ Wf1 + bf1)
        gemm_wmma<true, false, true><<<gF, blk, 0, stream>>>(
            h, Wf1 + oDF, bf1 + oF, nullptr, ff, GPT_FF, GPT_D);
        // x = x + ff @ Wf2 + bf2
        gemm_wmma<true, true, false><<<gD, blk, 0, stream>>>(
            ff, Wf2 + oDF, bf2 + oD, x, x, GPT_D, GPT_FF);
    }

    ln_kernel<<<NTOK, blk, 0, stream>>>(x, lnf_g, lnf_b, h);
    gemm_wmma<false, false, false><<<gV, blk, 0, stream>>>(
        h, W_lm, nullptr, nullptr, out, GPT_V, GPT_D);
}
